// TextGCN_29583734734915
// MI455X (gfx1250) — matrix operand order repaired and checked by reading, not compiled, not run on hardware
//
#include <hip/hip_runtime.h>

typedef __attribute__((ext_vector_type(4)))  _Float16 v4h;
typedef __attribute__((ext_vector_type(8)))  _Float16 v8h;
typedef __attribute__((ext_vector_type(16))) _Float16 v16h;
typedef __attribute__((ext_vector_type(8)))  float    v8f;

#define N_NODES 100000
#define E_EDGES 3200000
#define D_IN    512
#define D_HID   128
#define D_OUT   4

static __device__ __forceinline__ void atomicAddF(float* p, float v) {
  __hip_atomic_fetch_add(p, v, __ATOMIC_RELAXED, __HIP_MEMORY_SCOPE_AGENT);
}

__global__ void k_zero(float* __restrict__ p, int n) {
  int t = blockIdx.x * blockDim.x + threadIdx.x;
  if (t < n) p[t] = 0.0f;
}

// W1 [512][128] f32 -> Wt [128][512] f16 (so B fragments are contiguous in K)
__global__ void k_transpose_w1(const float* __restrict__ W1, _Float16* __restrict__ Wt) {
  int t = blockIdx.x * blockDim.x + threadIdx.x;   // 65536 threads
  int k = t >> 7;        // 0..511
  int c = t & 127;       // 0..127
  Wt[(size_t)c * D_IN + k] = (_Float16)W1[(size_t)k * D_HID + c];
}

__global__ void k_degree(const int* __restrict__ ei, float* __restrict__ deg) {
  int e = blockIdx.x * blockDim.x + threadIdx.x;   // exactly E
  int d = ei[E_EDGES + e];
  atomicAddF(&deg[d], 1.0f);
}

__global__ void k_dinv(float* __restrict__ deg) {
  int t = blockIdx.x * blockDim.x + threadIdx.x;
  if (t < N_NODES) deg[t] = rsqrtf(deg[t] + 1.0f);
}

// H1 = X @ W1 via v_wmma_f32_16x16x32_f16.
// Block = 256 threads = 8 waves; block computes 16 rows x 128 cols.
// Whole 16x512 A-panel staged to LDS (f16) once; then 16 barrier-free WMMA steps.
#define LDS_STRIDE 520   // 512 + 8 f16 pad; rows 1040 B apart (16B aligned)
__global__ __launch_bounds__(256) void k_gemm1_wmma(const float* __restrict__ X,
                                                    const _Float16* __restrict__ Wt,
                                                    float* __restrict__ H) {
  __shared__ _Float16 As[16 * LDS_STRIDE];   // ~16.6 KB of 320 KB WGP LDS

  const int wave = threadIdx.x >> 5;
  const int lane = threadIdx.x & 31;
  const int half = lane >> 4;   // lane group 0/1
  const int r    = lane & 15;   // row (A) / col (B,D) within tile
  const int rowBase = blockIdx.x * 16;   // 6250 blocks * 16 = 100000 exactly
  const int colBase = wave * 16;

  // ---- stage full 16x512 f32 panel -> f16 LDS, coalesced float4 loads ----
  const float4* Xp = (const float4*)(X + (size_t)rowBase * D_IN);  // 128 float4 per row
#pragma unroll
  for (int c = 0; c < 8; ++c) {
    int id  = c * 256 + threadIdx.x;   // 0..2047
    int row = id >> 7;                 // 0..15
    int c4  = id & 127;                // float4 index within row
    float4 xv = Xp[row * 128 + c4];
    v4h hv = { (_Float16)xv.x, (_Float16)xv.y, (_Float16)xv.z, (_Float16)xv.w };
    *(v4h*)(As + row * LDS_STRIDE + (c4 << 2)) = hv;
  }
  __syncthreads();

  // ---- 16 WMMA steps, no barriers ----
  v8f acc = {};
#pragma unroll
  for (int ks = 0; ks < 16; ++ks) {
    // A fragment (ISA 16-bit A 16x32 layout): lane r, K = [8h..8h+7] and [16+8h..23+8h]
    const _Float16* ar = As + r * LDS_STRIDE + ks * 32;
    v8h a0 = *(const v8h*)(ar + 8 * half);
    v8h a1 = *(const v8h*)(ar + 16 + 8 * half);
    v16h A = __builtin_shufflevector(a0, a1, 0,1,2,3,4,5,6,7,8,9,10,11,12,13,14,15);

    // B fragment: col = r, K = [8h..8h+7] and [16+8h..23+8h] in transposed W (16B aligned, L2-hot)
    const _Float16* bp = Wt + (size_t)(colBase + r) * D_IN + ks * 32;
    v8h b0 = *(const v8h*)(bp + 8 * half);
    v8h b1 = *(const v8h*)(bp + 16 + 8 * half);
    v16h B = __builtin_shufflevector(b0, b1, 0,1,2,3,4,5,6,7,8,9,10,11,12,13,14,15);

    acc = __builtin_amdgcn_wmma_f32_16x16x32_f16(false, A, false, B,
                                                 (short)0, acc, false, false);
  }

  // D layout: VGPR i -> row i+8*half, col = r
#pragma unroll
  for (int i = 0; i < 8; ++i) {
    int row = rowBase + i + 8 * half;
    H[(size_t)row * D_HID + colBase + r] = acc[i];
  }
}

// Edge scatter layer 1: 32 threads per edge, 4 feats per thread (coalesced float4 gather + FP atomics)
__global__ void k_agg1(const int* __restrict__ ei, const float* __restrict__ dinv,
                       const float* __restrict__ H, float* __restrict__ AGG) {
  int t  = blockIdx.x * blockDim.x + threadIdx.x;  // exactly E*32
  int e  = t >> 5;
  int f4 = (t & 31) << 2;
  int s = ei[e];
  int d = ei[E_EDGES + e];
  float w = dinv[s] * dinv[d];
  const float4 hv = *(const float4*)(H + (size_t)s * D_HID + f4);
  float* o = AGG + (size_t)d * D_HID + f4;
  atomicAddF(o + 0, hv.x * w);
  atomicAddF(o + 1, hv.y * w);
  atomicAddF(o + 2, hv.z * w);
  atomicAddF(o + 3, hv.w * w);
}

// h = relu(agg + h1 * dinv^2 + b1), written in place into AGG
__global__ void k_self_relu(const float* __restrict__ H1, const float* __restrict__ dinv,
                            const float* __restrict__ b1, float* __restrict__ AGG) {
  int t = blockIdx.x * blockDim.x + threadIdx.x;   // exactly N*128
  int n = t >> 7;
  int f = t & 127;
  float di = dinv[n];
  float v = AGG[t] + H1[t] * di * di + b1[f];
  AGG[t] = v > 0.0f ? v : 0.0f;
}

// Z = h @ W2  (100k x 128 x 4): thread per (n,c)
__global__ void k_gemm2(const float* __restrict__ Hc, const float* __restrict__ W2,
                        float* __restrict__ Z) {
  int t = blockIdx.x * blockDim.x + threadIdx.x;
  if (t >= N_NODES * D_OUT) return;
  int n = t >> 2;
  int c = t & 3;
  const float* hr = Hc + (size_t)n * D_HID;
  float a = 0.0f;
#pragma unroll 4
  for (int k = 0; k < D_HID; ++k) a += hr[k] * W2[k * D_OUT + c];
  Z[t] = a;
}

// Edge scatter layer 2: 4 threads per edge, accumulate directly into d_out
__global__ void k_agg2(const int* __restrict__ ei, const float* __restrict__ dinv,
                       const float* __restrict__ Z, float* __restrict__ OUT) {
  int t = blockIdx.x * blockDim.x + threadIdx.x;   // exactly E*4
  int e = t >> 2;
  int c = t & 3;
  int s = ei[e];
  int d = ei[E_EDGES + e];
  float w = dinv[s] * dinv[d];
  atomicAddF(&OUT[(size_t)d * D_OUT + c], Z[(size_t)s * D_OUT + c] * w);
}

__global__ void k_final(const float* __restrict__ Z, const float* __restrict__ dinv,
                        const float* __restrict__ b2, float* __restrict__ OUT) {
  int t = blockIdx.x * blockDim.x + threadIdx.x;
  if (t >= N_NODES * D_OUT) return;
  int n = t >> 2;
  int c = t & 3;
  float di = dinv[n];
  OUT[t] += Z[t] * di * di + b2[c];
}

extern "C" void kernel_launch(void* const* d_in, const int* in_sizes, int n_in,
                              void* d_out, int out_size, void* d_ws, size_t ws_size,
                              hipStream_t stream) {
  const float* X  = (const float*)d_in[0];
  const int*   EI = (const int*)d_in[1];     // [2][E], int32
  const float* W1 = (const float*)d_in[2];
  const float* B1 = (const float*)d_in[3];
  const float* W2 = (const float*)d_in[4];
  const float* B2 = (const float*)d_in[5];
  float* OUT = (float*)d_out;

  // Workspace layout (floats); all offsets 16B-aligned
  float* ws  = (float*)d_ws;
  float* deg = ws;                              // N            (becomes dinv in place)
  float* H1  = ws + 100000;                     // N*128
  float* AGG = ws + 100000 + 12800000;          // N*128
  float* Z   = ws + 25700000;                   // N*4
  _Float16* Wt = (_Float16*)(ws + 26100000);    // 128*512 f16

  const int B = 256;

  // init
  k_zero<<<(N_NODES + B - 1) / B, B, 0, stream>>>(deg, N_NODES);
  k_zero<<<(N_NODES * D_HID) / B, B, 0, stream>>>(AGG, N_NODES * D_HID);
  k_zero<<<(N_NODES * D_OUT + B - 1) / B, B, 0, stream>>>(OUT, N_NODES * D_OUT);

  // prep
  k_transpose_w1<<<(D_IN * D_HID) / B, B, 0, stream>>>(W1, Wt);
  k_degree<<<E_EDGES / B, B, 0, stream>>>(EI, deg);
  k_dinv<<<(N_NODES + B - 1) / B, B, 0, stream>>>(deg);

  // layer 1
  k_gemm1_wmma<<<N_NODES / 16, B, 0, stream>>>(X, Wt, H1);
  k_agg1<<<(E_EDGES * 32) / B, B, 0, stream>>>(EI, deg, H1, AGG);
  k_self_relu<<<(N_NODES * D_HID) / B, B, 0, stream>>>(H1, deg, B1, AGG);

  // layer 2
  k_gemm2<<<(N_NODES * D_OUT + B - 1) / B, B, 0, stream>>>(AGG, W2, Z);
  k_agg2<<<(E_EDGES * D_OUT) / B, B, 0, stream>>>(EI, deg, Z, OUT);
  k_final<<<(N_NODES * D_OUT + B - 1) / B, B, 0, stream>>>(Z, deg, B2, OUT);
}
